// CLFMv2_39436389712128
// MI455X (gfx1250) — matrix operand order repaired and checked by reading, not compiled, not run on hardware
//
#include <hip/hip_runtime.h>
#include <math.h>

// ---- problem dims (from reference) ----
#define B_    16
#define L_    12
#define NN    2048
#define C_    3
#define FD    256
#define HD    512
#define SD    256
#define OUTD  12
#define STEPS 4
#define M_    (B_ * NN)   // 32768 rows through the MLP stacks

typedef __attribute__((ext_vector_type(16))) __bf16 v16bf;
typedef __attribute__((ext_vector_type(8)))  float  v8f;
typedef __attribute__((ext_vector_type(4)))  int    v4i;

union FragU { v16bf v; uint4 q[2]; };

__device__ __forceinline__ unsigned short f2bf(float f) {
  unsigned u = __float_as_uint(f);
  u += 0x7FFFu + ((u >> 16) & 1u);            // round-to-nearest-even
  return (unsigned short)(u >> 16);
}
__device__ __forceinline__ float gelu_f(float v) {
  float c = v + 0.044715f * v * v * v;
  return 0.5f * v * (1.0f + tanhf(0.7978845608028654f * c));
}
__device__ __forceinline__ float sigmoid_f(float v) { return 1.0f / (1.0f + expf(-v)); }

// ---- CDNA5 async global->LDS copy (16B per lane), tracked by ASYNCcnt ----
__device__ __forceinline__ void async_copy16(const unsigned short* gsrc, unsigned short* ldst) {
#if __has_builtin(__builtin_amdgcn_global_load_async_to_lds_b128)
  typedef __attribute__((address_space(1))) v4i as1_v4i;   // prints as "__device__ int4*"
  typedef __attribute__((address_space(3))) v4i as3_v4i;   // prints as "__shared__ int4*"
  __builtin_amdgcn_global_load_async_to_lds_b128(
      (as1_v4i*)(uintptr_t)gsrc,
      (as3_v4i*)(unsigned)(uintptr_t)ldst,
      0, 0);
#else
  unsigned lds = (unsigned)(uintptr_t)ldst;
  asm volatile("global_load_async_to_lds_b128 %0, %1, off"
               :: "v"(lds), "v"(gsrc) : "memory");
#endif
}
template <int Nw>
__device__ __forceinline__ void wait_async() {
#if __has_builtin(__builtin_amdgcn_s_wait_asynccnt)
  __builtin_amdgcn_s_wait_asynccnt((unsigned short)Nw);
#else
  asm volatile("s_wait_asynccnt %0" :: "i"(Nw) : "memory");
#endif
}

// =====================================================================
// Generic bf16 WMMA GEMM:  C[z] = act( scale * (A[z] @ B[z]) + bias + rsv.*res[z] )
//   A: [M,K] bf16 row-major (lda).  B: bT=1 -> [Nn,K] (pre-transposed weights),
//   bT=0 -> [K,Nn] natural (e.g. the PDE field).  Dual fp32/bf16 outputs.
// Block = 256 thr = 8 waves; tile 128(M) x 64(N); K step 32.
// Double-buffered LDS; A (and transposed-B) tiles staged with async-to-LDS.
// =====================================================================
__global__ __launch_bounds__(256) void gemm_bf16_wmma(
    const unsigned short* __restrict__ A,  int lda, long long aBat,
    const unsigned short* __restrict__ Bm, int ldb, long long bBat, int bT,
    const float* __restrict__ bias,
    const float* __restrict__ scaleP,
    const float* __restrict__ res, int ldr, long long rBat,
    const float* __restrict__ resScaleVec,
    float* __restrict__ outF, int ldf, long long fBat,
    unsigned short* __restrict__ outH, int ldh, long long hBat,
    int M, int Nn, int K, int doGelu)
{
  // stride 48 ushorts -> 96B rows, keeps every 16B chunk aligned
  __shared__ __align__(16) unsigned short sA[2 * 128 * 48];
  __shared__ __align__(16) unsigned short sB[2 * 64 * 48];

  const int t  = threadIdx.x;
  const int bn = blockIdx.x, bm = blockIdx.y, z = blockIdx.z;
  A  += (long long)z * aBat;
  Bm += (long long)z * bBat;
  if (res)  res  += (long long)z * rBat;
  if (outF) outF += (long long)z * fBat;
  if (outH) outH += (long long)z * hBat;

  // staging indices
  const int ar = t >> 1;            // A: 128 rows x 2 halves of 16 bf16
  const int ac = (t & 1) * 16;
  const int br = t >> 2;            // B (bT=1): 64 rows x 4 chunks of 8
  const int bc = (t & 3) * 8;
  const int kk = t >> 3;            // B (bT=0): 32 k-rows x 8 col-chunks of 8
  const int nc = (t & 7) * 8;

  // wave tiling: 4(M) x 2(N) waves, each 32x32 via 2x2 WMMA accumulators
  const int lane  = t & 31;
  const int w     = t >> 5;
  const int mBase = (w & 3) * 32;
  const int nBase = (w >> 2) * 32;
  const int lm = lane & 15;
  const int kh = lane >> 4;

  v8f acc[2][2];
#pragma unroll
  for (int i = 0; i < 2; ++i)
#pragma unroll
    for (int j = 0; j < 2; ++j) acc[i][j] = (v8f){0,0,0,0,0,0,0,0};

  const unsigned short* gA = A + (size_t)(bm * 128 + ar) * lda + ac;
  unsigned short* lA = sA + ar * 48 + ac;
  const int nB1 = bn * 64 + br;                    // bT=1 staging row
  const int nB0 = bn * 64 + nc;                    // bT=0 staging col base

  // issue async staging for one K tile into buffer `buf`
  auto stageAsync = [&](int k0, int buf) {
    unsigned short* dA = lA + buf * (128 * 48);
    async_copy16(gA + k0,     dA);
    async_copy16(gA + k0 + 8, dA + 8);
    if (bT && nB1 < Nn)
      async_copy16(Bm + (size_t)nB1 * ldb + k0 + bc, sB + buf * (64 * 48) + br * 48 + bc);
  };
  // synchronous transpose-staging of natural-layout B (Laplacian GEMM path)
  auto stageBsync = [&](int k0, int buf) {
    if (nB0 < Nn) {
      uint4 x = *(const uint4*)(Bm + (size_t)(k0 + kk) * ldb + nB0);
      const unsigned short* px = (const unsigned short*)&x;
      unsigned short* db = sB + buf * (64 * 48);
#pragma unroll
      for (int j = 0; j < 8; ++j) db[(nc + j) * 48 + kk] = px[j];
    }
  };

  const int nTiles = K >> 5;
  stageAsync(0, 0);
  if (!bT) stageBsync(0, 0);

  for (int i = 0; i < nTiles; ++i) {
    const int nxt = i + 1;
    if (nxt < nTiles) {
      stageAsync(nxt << 5, nxt & 1);
      if (!bT) stageBsync(nxt << 5, nxt & 1);
      // allow the just-issued tile to stay in flight; tile i is then complete
      if (bT) wait_async<3>(); else wait_async<2>();
    } else {
      wait_async<0>();
    }
    __syncthreads();

    const unsigned short* cA = sA + (i & 1) * (128 * 48);
    const unsigned short* cB = sB + (i & 1) * (64 * 48);

    // ---- load fragments per ISA 7.12.2 layouts ----
    FragU a0, a1, b0, b1;
    {
      const unsigned short* p = cA + (mBase + lm) * 48 + 8 * kh;
      a0.q[0] = *(const uint4*)p;  a0.q[1] = *(const uint4*)(p + 16);
      p = cA + (mBase + 16 + lm) * 48 + 8 * kh;
      a1.q[0] = *(const uint4*)p;  a1.q[1] = *(const uint4*)(p + 16);
      p = cB + (nBase + lm) * 48 + 8 * kh;
      b0.q[0] = *(const uint4*)p;  b0.q[1] = *(const uint4*)(p + 16);
      p = cB + (nBase + 16 + lm) * 48 + 8 * kh;
      b1.q[0] = *(const uint4*)p;  b1.q[1] = *(const uint4*)(p + 16);
    }
    acc[0][0] = __builtin_amdgcn_wmma_f32_16x16x32_bf16(false, a0.v, false, b0.v, (short)0, acc[0][0], false, false);
    acc[0][1] = __builtin_amdgcn_wmma_f32_16x16x32_bf16(false, a0.v, false, b1.v, (short)0, acc[0][1], false, false);
    acc[1][0] = __builtin_amdgcn_wmma_f32_16x16x32_bf16(false, a1.v, false, b0.v, (short)0, acc[1][0], false, false);
    acc[1][1] = __builtin_amdgcn_wmma_f32_16x16x32_bf16(false, a1.v, false, b1.v, (short)0, acc[1][1], false, false);
    __syncthreads();
  }

  // ---- epilogue ----
  const float scale = scaleP ? *scaleP : 1.0f;
#pragma unroll
  for (int i = 0; i < 2; ++i)
#pragma unroll
    for (int j = 0; j < 2; ++j) {
      int col = bn * 64 + nBase + j * 16 + lm;
      if (col >= Nn) continue;
      float bv = bias ? bias[col] : 0.0f;
      float rv = resScaleVec ? resScaleVec[col] : 1.0f;
      int row0 = bm * 128 + mBase + i * 16 + kh * 8;
#pragma unroll
      for (int r = 0; r < 8; ++r) {
        int row = row0 + r;
        float v = acc[i][j][r] * scale + bv;
        if (res)    v += rv * res[(size_t)row * ldr + col];
        if (doGelu) v  = gelu_f(v);
        if (outF)   outF[(size_t)row * ldf + col] = v;
        if (outH)   outH[(size_t)row * ldh + col] = f2bf(v);
      }
    }
}

// =====================================================================
// Row LayerNorm (width W), optional post-GELU, dual fp32/bf16 outputs.
// =====================================================================
__global__ __launch_bounds__(256) void ln_rows(
    const float* __restrict__ X, const float* __restrict__ g, const float* __restrict__ be,
    float* __restrict__ outF, unsigned short* __restrict__ outH, int W, int postGelu)
{
  __shared__ float r1[256], r2[256];
  const int row = blockIdx.x, t = threadIdx.x;
  const float* xr = X + (size_t)row * W;
  float s = 0.f, s2 = 0.f;
  for (int j = t; j < W; j += 256) { float v = xr[j]; s += v; s2 += v * v; }
  r1[t] = s; r2[t] = s2; __syncthreads();
  for (int o = 128; o > 0; o >>= 1) {
    if (t < o) { r1[t] += r1[t + o]; r2[t] += r2[t + o]; }
    __syncthreads();
  }
  float mean = r1[0] / W;
  float var  = r2[0] / W - mean * mean;
  float rs   = rsqrtf(var + 1e-5f);
  for (int j = t; j < W; j += 256) {
    float v = (xr[j] - mean) * rs * g[j] + be[j];
    if (postGelu) v = gelu_f(v);
    if (outF) outF[(size_t)row * W + j] = v;
    if (outH) outH[(size_t)row * W + j] = f2bf(v);
  }
}

// ---- graph Laplacian prep ----
__global__ __launch_bounds__(256) void degree_kernel(const float* __restrict__ adj, float* __restrict__ dis) {
  __shared__ float r1[256];
  const int n = blockIdx.x, t = threadIdx.x;
  float s = 0.f;
  for (int m = t; m < NN; m += 256) s += adj[(size_t)n * NN + m];
  r1[t] = s; __syncthreads();
  for (int o = 128; o > 0; o >>= 1) { if (t < o) r1[t] += r1[t + o]; __syncthreads(); }
  if (t == 0) dis[n] = (r1[0] > 0.0f) ? rsqrtf(r1[0]) : 0.0f;
}
__global__ __launch_bounds__(256) void laplacian_kernel(const float* __restrict__ adj,
                                                        const float* __restrict__ dis,
                                                        unsigned short* __restrict__ LapH) {
  int idx = blockIdx.x * 256 + threadIdx.x;
  if (idx >= NN * NN) return;
  int n = idx / NN, m = idx - n * NN;
  float v = ((n == m) ? 1.0f : 0.0f) - dis[n] * adj[idx] * dis[m];
  LapH[idx] = f2bf(v);
}

// ---- weight convert fp32 [K,Nn] -> bf16 transposed [Nn,Kpad] (zero-padded K) ----
__global__ __launch_bounds__(256) void convert_w_t(const float* __restrict__ Wsrc,
                                                   unsigned short* __restrict__ Wt,
                                                   int K, int Nn, int Kpad) {
  int idx = blockIdx.x * 256 + threadIdx.x;
  if (idx >= Nn * Kpad) return;
  int n = idx / Kpad, k = idx - n * Kpad;
  Wt[idx] = (k < K) ? f2bf(Wsrc[(size_t)k * Nn + n]) : (unsigned short)0;
}

// ---- x [B,L,N,C] -> bf16 A0 [B*N, 64] (L*C=36 cols, zero-padded) ----
__global__ __launch_bounds__(256) void pack_x_kernel(const float* __restrict__ x,
                                                     unsigned short* __restrict__ A0) {
  int idx = blockIdx.x * 256 + threadIdx.x;
  if (idx >= M_ * 64) return;
  int row = idx >> 6, j = idx & 63;
  float v = 0.0f;
  if (j < L_ * C_) {
    int b = row / NN, n = row - b * NN;
    int l = j / C_, c = j - l * C_;
    v = x[(((size_t)b * L_ + l) * NN + n) * C_ + c];
  }
  A0[idx] = f2bf(v);
}

// ---- broadcast coords [N,HD] into concat cols [HD..2HD) of hcat [M,1024] ----
__global__ __launch_bounds__(256) void coords_fill_kernel(const float* __restrict__ coords,
                                                          unsigned short* __restrict__ hcat) {
  int idx = blockIdx.x * 256 + threadIdx.x;
  if (idx >= M_ * HD) return;
  int row = idx >> 9, j = idx & 511;
  int n = row % NN;
  hcat[(size_t)row * 1024 + HD + j] = f2bf(coords[(size_t)n * HD + j]);
}

// ---- scalars: [0]=alpha [1]=dcoef [2]=dt [3..3+SD)=A_disc ----
__global__ __launch_bounds__(256) void scalars_kernel(const float* __restrict__ logA,
                                                      const float* __restrict__ logdt,
                                                      const float* __restrict__ dco,
                                                      const float* __restrict__ alo,
                                                      float* __restrict__ sc) {
  int t = threadIdx.x;
  float dt = fminf(expf(logdt[0]), 1.0f);
  if (t == 0) { sc[0] = sigmoid_f(alo[0]); sc[1] = sigmoid_f(dco[0]); sc[2] = dt; }
  sc[3 + t] = expf(-expf(logA[t]) * dt);
}

__global__ __launch_bounds__(256) void zero_f32(float* __restrict__ p, int count) {
  int i = blockIdx.x * 256 + threadIdx.x;
  if (i < count) p[i] = 0.0f;
}

// ---- field += alpha*diff + (1-alpha)*neural ; dual write ----
__global__ __launch_bounds__(256) void euler_combine(const float* __restrict__ diff,
                                                     const float* __restrict__ neural,
                                                     const float* __restrict__ sc,
                                                     float* __restrict__ field,
                                                     unsigned short* __restrict__ fieldH) {
  int idx = blockIdx.x * 256 + threadIdx.x;
  if (idx >= M_ * FD) return;
  float alpha = sc[0];
  float f = field[idx] + alpha * diff[idx] + (1.0f - alpha) * neural[idx];
  field[idx] = f; fieldH[idx] = f2bf(f);
}

// ---- state = A_disc*state + Bu*dt ; dual write ----
__global__ __launch_bounds__(256) void ssm_state_kernel(const float* __restrict__ Bu,
                                                        const float* __restrict__ sc,
                                                        float* __restrict__ state,
                                                        unsigned short* __restrict__ stateH) {
  int idx = blockIdx.x * 256 + threadIdx.x;
  if (idx >= M_ * SD) return;
  int d = idx & (SD - 1);
  float s = sc[3 + d] * state[idx] + Bu[idx] * sc[2];
  state[idx] = s; stateH[idx] = f2bf(s);
}

// ---- pred [B*N, OUT] -> out [B, OUT, N, 1] ----
__global__ __launch_bounds__(256) void out_transpose(const float* __restrict__ pred,
                                                     float* __restrict__ out) {
  int idx = blockIdx.x * 256 + threadIdx.x;
  if (idx >= B_ * OUTD * NN) return;
  int n = idx % NN; int rest = idx / NN;
  int o = rest % OUTD; int b = rest / OUTD;
  out[idx] = pred[((size_t)b * NN + n) * OUTD + o];
}

// =====================================================================
extern "C" void kernel_launch(void* const* d_in, const int* in_sizes, int n_in,
                              void* d_out, int out_size, void* d_ws, size_t ws_size,
                              hipStream_t stream) {
  (void)in_sizes; (void)n_in; (void)out_size; (void)ws_size;
  // ---- flattened input order (setup_inputs dict, depth-first) ----
  const float* x        = (const float*)d_in[0];
  const float* adj      = (const float*)d_in[1];
  const float* encPw    = (const float*)d_in[2];
  const float* encPb    = (const float*)d_in[3];
  const float* encCoord = (const float*)d_in[4];
  const float* e0w1 = (const float*)d_in[5];  const float* e0b1 = (const float*)d_in[6];
  const float* e0w2 = (const float*)d_in[7];  const float* e0b2 = (const float*)d_in[8];
  const float* e0g  = (const float*)d_in[9];  const float* e0be = (const float*)d_in[10];
  const float* e1w1 = (const float*)d_in[11]; const float* e1b1 = (const float*)d_in[12];
  const float* e1w2 = (const float*)d_in[13]; const float* e1b2 = (const float*)d_in[14];
  const float* e1g  = (const float*)d_in[15]; const float* e1be = (const float*)d_in[16];
  const float* tfW  = (const float*)d_in[17]; const float* tfB  = (const float*)d_in[18];
  const float* p0w  = (const float*)d_in[19]; const float* p0b  = (const float*)d_in[20];
  const float* p0g  = (const float*)d_in[21]; const float* p0be = (const float*)d_in[22];
  const float* p1w  = (const float*)d_in[23]; const float* p1b  = (const float*)d_in[24];
  const float* p1g  = (const float*)d_in[25]; const float* p1be = (const float*)d_in[26];
  const float* poW  = (const float*)d_in[27]; const float* poB  = (const float*)d_in[28];
  const float* logA = (const float*)d_in[29];
  const float* sBw  = (const float*)d_in[30]; const float* sBb  = (const float*)d_in[31];
  const float* sCw  = (const float*)d_in[32]; const float* sCb  = (const float*)d_in[33];
  const float* sD   = (const float*)d_in[34]; const float* logdt= (const float*)d_in[35];
  const float* dPw  = (const float*)d_in[36]; const float* dPb  = (const float*)d_in[37];
  const float* decCoord = (const float*)d_in[38];
  const float* d0w1 = (const float*)d_in[39]; const float* d0b1 = (const float*)d_in[40];
  const float* d0w2 = (const float*)d_in[41]; const float* d0b2 = (const float*)d_in[42];
  const float* d0g  = (const float*)d_in[43]; const float* d0be = (const float*)d_in[44];
  const float* d1w1 = (const float*)d_in[45]; const float* d1b1 = (const float*)d_in[46];
  const float* d1w2 = (const float*)d_in[47]; const float* d1b2 = (const float*)d_in[48];
  const float* d1g  = (const float*)d_in[49]; const float* d1be = (const float*)d_in[50];
  const float* doW  = (const float*)d_in[51]; const float* doB  = (const float*)d_in[52];
  const float* dco  = (const float*)d_in[53]; const float* alo  = (const float*)d_in[54];
  float* outP = (float*)d_out;

  // ---- workspace bump allocator ----
  char* wsb = (char*)d_ws; size_t off = 0;
  auto carve = [&](size_t bytes) -> void* {
    void* p = wsb + off; off = (off + bytes + 255) & ~(size_t)255; return p;
  };
  float*          dis_   = (float*)carve(NN * 4);
  float*          sc_    = (float*)carve(1024);
  unsigned short* Lap    = (unsigned short*)carve((size_t)NN * NN * 2);
  unsigned short* A0     = (unsigned short*)carve((size_t)M_ * 64 * 2);
  unsigned short* hcat   = (unsigned short*)carve((size_t)M_ * 1024 * 2);
  float*          Tf     = (float*)carve((size_t)M_ * HD * 4);
  unsigned short* Th     = (unsigned short*)carve((size_t)M_ * HD * 2);
  float*          Rf     = (float*)carve((size_t)M_ * HD * 4);
  unsigned short* Rh     = (unsigned short*)carve((size_t)M_ * HD * 2);
  float*          fieldF = (float*)carve((size_t)M_ * FD * 4);
  unsigned short* fieldH = (unsigned short*)carve((size_t)M_ * FD * 2);
  float*          diffB  = (float*)carve((size_t)M_ * FD * 4);
  float*          neur   = (float*)carve((size_t)M_ * FD * 4);
  float*          BuB    = (float*)carve((size_t)M_ * SD * 4);
  float*          state  = (float*)carve((size_t)M_ * SD * 4);
  unsigned short* stateH = (unsigned short*)carve((size_t)M_ * SD * 2);
  float*          pred   = (float*)carve((size_t)M_ * OUTD * 4);
  // bf16 transposed weights [Nn][Kpad]
  auto carveW = [&](int Nn, int Kp) { return (unsigned short*)carve((size_t)Nn * Kp * 2); };
  unsigned short* WencP = carveW(HD, 64);
  unsigned short* We0w1 = carveW(HD, 1024);  unsigned short* We0w2 = carveW(HD, HD);
  unsigned short* We1w1 = carveW(HD, HD);    unsigned short* We1w2 = carveW(HD, HD);
  unsigned short* Wtf   = carveW(FD, HD);
  unsigned short* Wp0   = carveW(HD, FD);    unsigned short* Wp1   = carveW(HD, HD);
  unsigned short* Wpo   = carveW(FD, HD);
  unsigned short* WsB   = carveW(SD, FD);    unsigned short* WsC   = carveW(FD, SD);
  unsigned short* WdP   = carveW(HD, FD);
  unsigned short* Wd0w1 = carveW(HD, 1024);  unsigned short* Wd0w2 = carveW(HD, HD);
  unsigned short* Wd1w1 = carveW(HD, HD);    unsigned short* Wd1w2 = carveW(HD, HD);
  unsigned short* Wdo   = carveW(OUTD, HD);

  auto ew = [&](int count) { return dim3((count + 255) / 256); };
  auto cw = [&](const float* src, unsigned short* dst, int K, int Nn, int Kp) {
    convert_w_t<<<ew(Nn * Kp), 256, 0, stream>>>(src, dst, K, Nn, Kp);
  };
  auto gemm = [&](const unsigned short* Aa, int lda, long long aB,
                  const unsigned short* Bb, int ldb, long long bB, int bT,
                  const float* bias, const float* scaleP,
                  const float* res, int ldr, long long rB, const float* rsv,
                  float* oF, int ldf, long long fB,
                  unsigned short* oH, int ldh, long long hB,
                  int Ml, int Nn, int K, int gelu, int batches) {
    dim3 grid((Nn + 63) / 64, Ml / 128, batches);
    gemm_bf16_wmma<<<grid, 256, 0, stream>>>(Aa, lda, aB, Bb, ldb, bB, bT, bias, scaleP,
                                             res, ldr, rB, rsv, oF, ldf, fB, oH, ldh, hB,
                                             Ml, Nn, K, gelu);
  };

  // ---- prep ----
  scalars_kernel<<<1, 256, 0, stream>>>(logA, logdt, dco, alo, sc_);
  degree_kernel<<<NN, 256, 0, stream>>>(adj, dis_);
  laplacian_kernel<<<ew(NN * NN), 256, 0, stream>>>(adj, dis_, Lap);
  pack_x_kernel<<<ew(M_ * 64), 256, 0, stream>>>(x, A0);
  zero_f32<<<ew(M_ * SD), 256, 0, stream>>>(state, M_ * SD);
  cw(encPw, WencP, 36, HD, 64);
  cw(e0w1, We0w1, 1024, HD, 1024);  cw(e0w2, We0w2, HD, HD, HD);
  cw(e1w1, We1w1, HD, HD, HD);      cw(e1w2, We1w2, HD, HD, HD);
  cw(tfW, Wtf, HD, FD, HD);
  cw(p0w, Wp0, FD, HD, FD);         cw(p1w, Wp1, HD, HD, HD);
  cw(poW, Wpo, HD, FD, HD);
  cw(sBw, WsB, FD, SD, FD);         cw(sCw, WsC, SD, FD, SD);
  cw(dPw, WdP, FD, HD, FD);
  cw(d0w1, Wd0w1, 1024, HD, 1024);  cw(d0w2, Wd0w2, HD, HD, HD);
  cw(d1w1, Wd1w1, HD, HD, HD);      cw(d1w2, Wd1w2, HD, HD, HD);
  cw(doW, Wdo, HD, OUTD, HD);

  // ---- encoder ----
  coords_fill_kernel<<<ew(M_ * HD), 256, 0, stream>>>(encCoord, hcat);
  gemm(A0, 64, 0, WencP, 64, 0, 1, encPb, nullptr, nullptr, 0, 0, nullptr,
       nullptr, 0, 0, hcat, 1024, 0, M_, HD, 64, 0, 1);                       // h -> hcat[:, :512]
  gemm(hcat, 1024, 0, We0w1, 1024, 0, 1, e0b1, nullptr, nullptr, 0, 0, nullptr,
       nullptr, 0, 0, Th, HD, 0, M_, HD, 1024, 1, 1);                          // gelu(hcat@w1+b1)
  gemm(Th, HD, 0, We0w2, HD, 0, 1, e0b2, nullptr, nullptr, 0, 0, nullptr,
       Tf, HD, 0, nullptr, 0, 0, M_, HD, HD, 0, 1);                            // @w2+b2 (no res)
  ln_rows<<<M_, 256, 0, stream>>>(Tf, e0g, e0be, Rf, Rh, HD, 0);
  gemm(Rh, HD, 0, We1w1, HD, 0, 1, e1b1, nullptr, nullptr, 0, 0, nullptr,
       nullptr, 0, 0, Th, HD, 0, M_, HD, HD, 1, 1);
  gemm(Th, HD, 0, We1w2, HD, 0, 1, e1b2, nullptr, Rf, HD, 0, nullptr,
       Tf, HD, 0, nullptr, 0, 0, M_, HD, HD, 0, 1);                            // + residual
  ln_rows<<<M_, 256, 0, stream>>>(Tf, e1g, e1be, nullptr, Rh, HD, 0);
  gemm(Rh, HD, 0, Wtf, HD, 0, 1, tfB, nullptr, nullptr, 0, 0, nullptr,
       fieldF, FD, 0, fieldH, FD, 0, M_, FD, HD, 0, 1);                        // field

  // ---- PDE evolution + SSM refine ----
  const long long bat = (long long)NN * FD;
  for (int s = 0; s < STEPS; ++s) {
    // diff = dcoef * (Lmat @ field[b]) : batched over B_, shared A
    gemm(Lap, NN, 0, fieldH, FD, bat, 0, nullptr, sc_ + 1, nullptr, 0, 0, nullptr,
         diffB, FD, bat, nullptr, 0, 0, NN, FD, NN, 0, B_);
    // neural path
    gemm(fieldH, FD, 0, Wp0, FD, 0, 1, p0b, nullptr, nullptr, 0, 0, nullptr,
         Tf, HD, 0, nullptr, 0, 0, M_, HD, FD, 0, 1);
    ln_rows<<<M_, 256, 0, stream>>>(Tf, p0g, p0be, nullptr, Th, HD, 1);        // gelu(ln(.))
    gemm(Th, HD, 0, Wp1, HD, 0, 1, p1b, nullptr, nullptr, 0, 0, nullptr,
         Tf, HD, 0, nullptr, 0, 0, M_, HD, HD, 0, 1);
    ln_rows<<<M_, 256, 0, stream>>>(Tf, p1g, p1be, nullptr, Th, HD, 1);
    gemm(Th, HD, 0, Wpo, HD, 0, 1, poB, nullptr, nullptr, 0, 0, nullptr,
         neur, FD, 0, nullptr, 0, 0, M_, FD, HD, 0, 1);
    euler_combine<<<ew(M_ * FD), 256, 0, stream>>>(diffB, neur, sc_, fieldF, fieldH);
    // SSM: Bu = field@Bw + b; state = A*state + Bu*dt; field = state@Cw + b + D*field
    gemm(fieldH, FD, 0, WsB, FD, 0, 1, sBb, nullptr, nullptr, 0, 0, nullptr,
         BuB, SD, 0, nullptr, 0, 0, M_, SD, FD, 0, 1);
    ssm_state_kernel<<<ew(M_ * SD), 256, 0, stream>>>(BuB, sc_, state, stateH);
    gemm(stateH, SD, 0, WsC, SD, 0, 1, sCb, nullptr, fieldF, FD, 0, sD,
         fieldF, FD, 0, fieldH, FD, 0, M_, FD, SD, 0, 1);
  }

  // ---- decoder ----
  coords_fill_kernel<<<ew(M_ * HD), 256, 0, stream>>>(decCoord, hcat);
  gemm(fieldH, FD, 0, WdP, FD, 0, 1, dPb, nullptr, nullptr, 0, 0, nullptr,
       nullptr, 0, 0, hcat, 1024, 0, M_, HD, FD, 0, 1);
  gemm(hcat, 1024, 0, Wd0w1, 1024, 0, 1, d0b1, nullptr, nullptr, 0, 0, nullptr,
       nullptr, 0, 0, Th, HD, 0, M_, HD, 1024, 1, 1);
  gemm(Th, HD, 0, Wd0w2, HD, 0, 1, d0b2, nullptr, nullptr, 0, 0, nullptr,
       Tf, HD, 0, nullptr, 0, 0, M_, HD, HD, 0, 1);
  ln_rows<<<M_, 256, 0, stream>>>(Tf, d0g, d0be, Rf, Rh, HD, 0);
  gemm(Rh, HD, 0, Wd1w1, HD, 0, 1, d1b1, nullptr, nullptr, 0, 0, nullptr,
       nullptr, 0, 0, Th, HD, 0, M_, HD, HD, 1, 1);
  gemm(Th, HD, 0, Wd1w2, HD, 0, 1, d1b2, nullptr, Rf, HD, 0, nullptr,
       Tf, HD, 0, nullptr, 0, 0, M_, HD, HD, 0, 1);
  ln_rows<<<M_, 256, 0, stream>>>(Tf, d1g, d1be, nullptr, Rh, HD, 0);
  gemm(Rh, HD, 0, Wdo, HD, 0, 1, doB, nullptr, nullptr, 0, 0, nullptr,
       pred, OUTD, 0, nullptr, 0, 0, M_, OUTD, HD, 0, 1);
  out_transpose<<<ew(B_ * OUTD * NN), 256, 0, stream>>>(pred, outP);
}
